// gaudi_MambaMixer_36996848288283
// MI455X (gfx1250) — matrix operand order repaired and checked by reading, not compiled, not run on hardware
//
#include <hip/hip_runtime.h>

// ---------------- problem constants ----------------
#define BB 2
#define LL 1024
#define HH 2048
#define DD 4096
#define NN 16
#define RR 128
#define KK 4
#define MM (BB * LL)          // 2048 tokens
#define P2D (2 * DD)          // 8192
#define NXPAD 256             // x_proj weight rows padded to multiple of 128

// ---------------- types ----------------
typedef __attribute__((ext_vector_type(16))) __bf16        v16bf;
typedef __attribute__((ext_vector_type(16))) unsigned short v16u;
typedef __attribute__((ext_vector_type(8)))  float          v8f;

__device__ __forceinline__ unsigned short f2bf(float f) {
  unsigned int u = __builtin_bit_cast(unsigned int, f);
  unsigned int r = u + 0x7fffu + ((u >> 16) & 1u);   // round-to-nearest-even
  return (unsigned short)(r >> 16);
}

__device__ __forceinline__ float siluf(float x) {
  return x * (1.0f / (1.0f + __expf(-x)));
}

__device__ __forceinline__ float softplusf(float x) {
  return (x > 20.0f) ? x : log1pf(__expf(x));
}

// low 32 bits of a generic LDS pointer == DS byte address (aperture layout)
__device__ __forceinline__ unsigned lds_off(const void* p) {
  return (unsigned)(unsigned long long)p;
}

// async gather: per-lane 16B global -> LDS, tracked by ASYNCcnt (no VGPR transit)
__device__ __forceinline__ void async_g2l_b128(unsigned loff, const void* gaddr) {
  asm volatile("global_load_async_to_lds_b128 %0, %1, off"
               :: "v"(loff), "v"(gaddr) : "memory");
}

__device__ __forceinline__ void wait_async_le0() {
#if __has_builtin(__builtin_amdgcn_s_wait_asynccnt)
  __builtin_amdgcn_s_wait_asynccnt(0);
#else
  asm volatile("s_wait_asynccnt 0x0" ::: "memory");
#endif
}

// async loads complete in order per wave: <=4 outstanding => previous tile landed
__device__ __forceinline__ void wait_async_le4() {
#if __has_builtin(__builtin_amdgcn_s_wait_asynccnt)
  __builtin_amdgcn_s_wait_asynccnt(4);
#else
  asm volatile("s_wait_asynccnt 0x4" ::: "memory");
#endif
}

// ---------------- f32 -> bf16 convert ----------------
__global__ __launch_bounds__(256) void cvt_bf16_kernel(const float* __restrict__ in,
                                                       unsigned short* __restrict__ out,
                                                       int n) {
  int i = blockIdx.x * 256 + threadIdx.x;
  if (i < n) out[i] = f2bf(in[i]);
}

// convert with zero tail (pads weight matrices so GEMM tiles never go OOB)
__global__ __launch_bounds__(256) void cvt_bf16_pad_kernel(const float* __restrict__ in,
                                                           unsigned short* __restrict__ out,
                                                           int nvalid, int ntotal) {
  int i = blockIdx.x * 256 + threadIdx.x;
  if (i < ntotal) out[i] = (i < nvalid) ? f2bf(in[i]) : (unsigned short)0;
}

// ---------------- WMMA fragment builders (CDNA5 wave32 layouts) ----------------
// A 16x32 bf16: lanes 0-15 -> M rows (K half 0), lanes 16-31 -> same rows, K+8.
// element e (VGPR v=e>>1): k = (v>=4?16:0) + half*8 + 2*(v&3) + (e&1)
__device__ __forceinline__ v16bf load_a_frag(const unsigned short* __restrict__ As,
                                             int mBase, int lane) {
  const int row  = mBase + (lane & 15);
  const int half = (lane >> 4) & 1;
  v16u t;
#pragma unroll
  for (int e = 0; e < 16; ++e) {
    const int v = e >> 1;
    const int k = ((v >= 4) ? 16 : 0) + half * 8 + 2 * (v & 3) + (e & 1);
    t[e] = As[row * 32 + k];
  }
  return __builtin_bit_cast(v16bf, t);
}

// B 32x16 bf16 (col-per-lane): lane n = lane&15, half = lane>>4; elem e -> k = (e>=8?16:0) + half*8 + (e&7)
__device__ __forceinline__ v16bf load_b_frag(const unsigned short* __restrict__ Bs,
                                             int nBase, int lane) {
  const int col  = nBase + (lane & 15);
  const int half = (lane >> 4) & 1;
  v16u t;
#pragma unroll
  for (int e = 0; e < 16; ++e) t[e] = Bs[col * 32 + half * 8 + (e < 8 ? e : e + 8)];
  return __builtin_bit_cast(v16bf, t);
}

// ---------------- bf16 WMMA GEMM (double-buffered async-LDS pipeline) ----------------
// C[M,N] = A[M,K] (bf16 row-major) * Bw[N,K]^T (bf16 row-major weight, N padded to 128)
// block tile 128x128, 8 waves = 4(M) x 2(N), wave tile 32x64 (2x4 WMMA), BK = 32
// EPI 0: row-major f32  C[m*ldc + n]
// EPI 1: transpose      C[(m/Lc)*ldc*Lc + n*Lc + m%Lc]        (ldc = P dimension)
// EPI 2: EPI1 + bias[n] + softplus                              (dt_proj)
// EPI 3: EPI0 + bf16 copy of cols n<RR into aux[m*RR+n]         (x_proj)
template <int EPI>
__global__ __launch_bounds__(256) void gemm_bf16_wmma(
    const unsigned short* __restrict__ A, const unsigned short* __restrict__ Bw,
    float* __restrict__ C, int Mdim, int Nreal, int Kd,
    const float* __restrict__ bias, unsigned short* __restrict__ aux,
    int Lc, int ldc) {
  __shared__ unsigned short As[2][128 * 32];  // 2 x 8 KB
  __shared__ unsigned short Bs[2][128 * 32];  // 2 x 8 KB

  const int tid  = threadIdx.x;
  const int wave = tid >> 5;
  const int lane = tid & 31;
  const int wm   = wave >> 1;   // 0..3 (M, 32 rows each)
  const int wn   = wave & 1;    // 0..1 (N, 64 cols each)
  const int m0   = blockIdx.y * 128;
  const int n0   = blockIdx.x * 128;

  const int T = Kd >> 5;  // K tiles of 32

  // each call issues exactly 4 async ops per wave (uniform: inputs are padded)
  auto issue = [&](int t, int bsel) {
    const int k0 = t << 5;
#pragma unroll
    for (int i = 0; i < 2; ++i) {
      const int c   = tid + i * 256;
      const int row = c >> 2;
      const int cc  = (c & 3) * 8;
      async_g2l_b128(lds_off(&As[bsel][row * 32 + cc]),
                     A + (size_t)(m0 + row) * Kd + k0 + cc);
      async_g2l_b128(lds_off(&Bs[bsel][row * 32 + cc]),
                     Bw + (size_t)(n0 + row) * Kd + k0 + cc);
    }
  };

  v8f acc[2][4] = {};

  issue(0, 0);
  if (T > 1) issue(1, 1);

  for (int t = 0; t < T; ++t) {
    if (t + 1 < T) wait_async_le4();  // tile t landed; tile t+1 still streaming
    else           wait_async_le0();
    __syncthreads();

    const unsigned short* Ab = As[t & 1];
    const unsigned short* Bb = Bs[t & 1];

    v16bf af[2], bfm[4];
    af[0] = load_a_frag(Ab, wm * 32,      lane);
    af[1] = load_a_frag(Ab, wm * 32 + 16, lane);
#pragma unroll
    for (int j = 0; j < 4; ++j)
      bfm[j] = load_b_frag(Bb, wn * 64 + j * 16, lane);

#pragma unroll
    for (int i = 0; i < 2; ++i)
#pragma unroll
      for (int j = 0; j < 4; ++j)
        acc[i][j] = __builtin_amdgcn_wmma_f32_16x16x32_bf16(
            false, af[i], false, bfm[j], (short)0, acc[i][j], false, false);

    __syncthreads();                   // all waves done reading buffer t&1
    if (t + 2 < T) issue(t + 2, t & 1);  // overwrite it behind the compute
    if (t + 4 < T)                     // nudge far weight tile toward L2
      __builtin_prefetch(Bw + (size_t)(n0 + (tid >> 2)) * Kd + ((t + 4) << 5), 0, 1);
  }

  // epilogue: lane holds column n fixed; 8 elements are 8 consecutive rows m
#pragma unroll
  for (int i = 0; i < 2; ++i)
#pragma unroll
    for (int j = 0; j < 4; ++j) {
      const int n  = n0 + wn * 64 + j * 16 + (lane & 15);
      const int mb = m0 + wm * 32 + i * 16 + ((lane >> 4) & 1) * 8;
      if (n >= Nreal) continue;
      if constexpr (EPI == 0 || EPI == 3) {
#pragma unroll
        for (int r = 0; r < 8; ++r) {
          const float v = acc[i][j][r];
          C[(size_t)(mb + r) * ldc + n] = v;
          if constexpr (EPI == 3) {
            if (n < RR) aux[(size_t)(mb + r) * RR + n] = f2bf(v);
          }
        }
      } else {
        const int b = mb / Lc;
        const int l = mb % Lc;
        float* dst  = C + ((size_t)b * ldc + n) * (size_t)Lc + l;  // contiguous in L
#pragma unroll
        for (int r = 0; r < 8; ++r) {
          float v = acc[i][j][r];
          if constexpr (EPI == 2) v = softplusf(v + bias[n]);
          dst[r] = v;
        }
      }
    }
}

// ---------------- depthwise causal conv (K=4) + SiLU ----------------
__global__ __launch_bounds__(256) void conv_silu_kernel(
    const float* __restrict__ proj, const float* __restrict__ cw,
    const float* __restrict__ cb, float* __restrict__ u_scan,
    unsigned short* __restrict__ u_tok) {
  const size_t idx = (size_t)blockIdx.x * 256 + threadIdx.x;
  const size_t total = (size_t)BB * DD * LL;
  if (idx >= total) return;
  const int l = (int)(idx % LL);
  const int d = (int)((idx / LL) % DD);
  const int b = (int)(idx / ((size_t)LL * DD));
  const float* up = proj + ((size_t)b * P2D + d) * LL;
  float acc = cb[d];
#pragma unroll
  for (int k = 0; k < KK; ++k) {
    const int li = l - (KK - 1) + k;
    if (li >= 0) acc += up[li] * cw[d * KK + k];
  }
  const float s = siluf(acc);
  u_scan[idx] = s;
  u_tok[((size_t)(b * LL + l)) * DD + d] = f2bf(s);
}

// ---------------- selective scan ----------------
__global__ __launch_bounds__(256) void scan_kernel(
    const float* __restrict__ dtb,  // [B,D,L]
    const float* __restrict__ u,    // [B,D,L]
    const float* __restrict__ ssm,  // [M,160]: [:,128:144]=B, [:,144:160]=C
    const float* __restrict__ A_log,// [D,N]
    float* __restrict__ y) {        // [B,D,L]
  const int DPB = 256;
  const int b = blockIdx.x / (DD / DPB);
  const int d = (blockIdx.x % (DD / DPB)) * DPB + threadIdx.x;

  __shared__ float BC[64][32];  // [step][0..15]=B_t, [16..31]=C_t

  float Areg[NN], st[NN];
#pragma unroll
  for (int n = 0; n < NN; ++n) {
    Areg[n] = -__expf(A_log[d * NN + n]);
    st[n] = 0.0f;
  }
  const float* dtp = dtb + ((size_t)b * DD + d) * LL;
  const float* up  = u   + ((size_t)b * DD + d) * LL;
  float*       yp  = y   + ((size_t)b * DD + d) * LL;

  for (int c = 0; c < LL; c += 64) {
    __syncthreads();
#pragma unroll
    for (int i = 0; i < 8; ++i) {
      const int f = threadIdx.x * 8 + i;  // 0..2047
      const int s = f >> 5, w = f & 31;
      BC[s][w] = ssm[(size_t)(b * LL + c + s) * (RR + 2 * NN) + RR + w];
    }
    __syncthreads();
    for (int s = 0; s < 64; ++s) {
      const float dt  = dtp[c + s];
      const float dtu = dt * up[c + s];
      float yv = 0.0f;
#pragma unroll
      for (int n = 0; n < NN; ++n) {
        const float dA = __expf(Areg[n] * dt);
        st[n] = dA * st[n] + dtu * BC[s][n];
        yv += st[n] * BC[s][16 + n];
      }
      yp[c + s] = yv;
    }
  }
}

// ---------------- gating: y = (y + u*Dp) * silu(gate); token-major bf16 ----------------
__global__ __launch_bounds__(256) void gate_kernel(
    const float* __restrict__ y, const float* __restrict__ u,
    const float* __restrict__ Dp, const float* __restrict__ proj,
    unsigned short* __restrict__ ytok) {
  const size_t idx = (size_t)blockIdx.x * 256 + threadIdx.x;
  const size_t total = (size_t)BB * DD * LL;
  if (idx >= total) return;
  const int l = (int)(idx % LL);
  const int d = (int)((idx / LL) % DD);
  const int b = (int)(idx / ((size_t)LL * DD));
  const float g = proj[((size_t)b * P2D + DD + d) * LL + l];
  const float v = (y[idx] + u[idx] * Dp[d]) * siluf(g);
  ytok[((size_t)(b * LL + l)) * DD + d] = f2bf(v);
}

// ---------------- host ----------------
extern "C" void kernel_launch(void* const* d_in, const int* in_sizes, int n_in,
                              void* d_out, int out_size, void* d_ws, size_t ws_size,
                              hipStream_t stream) {
  const float* hs      = (const float*)d_in[0];  // [B,L,H]
  const float* w_in    = (const float*)d_in[1];  // [2D,H]
  const float* conv_w  = (const float*)d_in[2];  // [D,1,K]
  const float* conv_b  = (const float*)d_in[3];  // [D]
  const float* w_x     = (const float*)d_in[4];  // [R+2N, D]
  const float* w_dt    = (const float*)d_in[5];  // [D,R]
  const float* b_dt    = (const float*)d_in[6];  // [D]
  const float* A_log   = (const float*)d_in[7];  // [D,N]
  const float* D_param = (const float*)d_in[8];  // [D]
  const float* w_out   = (const float*)d_in[9];  // [H,D]
  float* out = (float*)d_out;                    // [B,L,H]

  char* p = (char*)d_ws;
  auto alloc = [&](size_t bytes) {
    void* r = (void*)p;
    p += (bytes + 255) & ~(size_t)255;
    return r;
  };

  unsigned short* hs_bf  = (unsigned short*)alloc((size_t)MM * HH * 2);
  unsigned short* win_bf = (unsigned short*)alloc((size_t)P2D * HH * 2);
  unsigned short* wx_bf  = (unsigned short*)alloc((size_t)NXPAD * DD * 2);  // padded
  unsigned short* wdt_bf = (unsigned short*)alloc((size_t)DD * RR * 2);
  unsigned short* wo_bf  = (unsigned short*)alloc((size_t)HH * DD * 2);
  float* proj    = (float*)alloc((size_t)BB * P2D * LL * 4);  // [B,2D,L]
  float* u_scan  = (float*)alloc((size_t)BB * DD * LL * 4);   // [B,D,L]
  unsigned short* u_tok = (unsigned short*)alloc((size_t)MM * DD * 2);
  float* ssm     = (float*)alloc((size_t)MM * (RR + 2 * NN) * 4);
  unsigned short* ts_bf = (unsigned short*)alloc((size_t)MM * RR * 2);
  float* dtbuf   = (float*)alloc((size_t)BB * DD * LL * 4);   // [B,D,L]
  float* ybuf    = (float*)alloc((size_t)BB * DD * LL * 4);   // [B,D,L]
  unsigned short* y_tok = (unsigned short*)alloc((size_t)MM * DD * 2);

  auto cvt = [&](const float* src, unsigned short* dst, int n) {
    cvt_bf16_kernel<<<(n + 255) / 256, 256, 0, stream>>>(src, dst, n);
  };
  cvt(hs,    hs_bf,  MM * HH);
  cvt(w_in,  win_bf, P2D * HH);
  cvt(w_dt,  wdt_bf, DD * RR);
  cvt(w_out, wo_bf,  HH * DD);
  // x_proj weights: 160 real rows, zero-padded to 256 so B tiles never go OOB
  {
    const int nvalid = (RR + 2 * NN) * DD, ntotal = NXPAD * DD;
    cvt_bf16_pad_kernel<<<(ntotal + 255) / 256, 256, 0, stream>>>(w_x, wx_bf,
                                                                  nvalid, ntotal);
  }

  // 1) in_proj: proj[B,2D,L] = hs[M,H] x w_in[2D,H]^T (transpose epilogue)
  gemm_bf16_wmma<1><<<dim3(P2D / 128, MM / 128), 256, 0, stream>>>(
      hs_bf, win_bf, proj, MM, P2D, HH, nullptr, nullptr, LL, P2D);

  // 2) depthwise causal conv + SiLU
  {
    const size_t total = (size_t)BB * DD * LL;
    conv_silu_kernel<<<(unsigned)((total + 255) / 256), 256, 0, stream>>>(
        proj, conv_w, conv_b, u_scan, u_tok);
  }

  // 3) x_proj: ssm[M,160] = u_tok[M,D] x w_x[160,D]^T (+ bf16 copy of dt cols)
  gemm_bf16_wmma<3><<<dim3(NXPAD / 128, MM / 128), 256, 0, stream>>>(
      u_tok, wx_bf, ssm, MM, RR + 2 * NN, DD, nullptr, ts_bf, LL, RR + 2 * NN);

  // 4) dt_proj: dt[B,D,L] = softplus(ts[M,R] x w_dt[D,R]^T + b_dt) (transpose epi)
  gemm_bf16_wmma<2><<<dim3(DD / 128, MM / 128), 256, 0, stream>>>(
      ts_bf, wdt_bf, dtbuf, MM, DD, RR, b_dt, nullptr, LL, DD);

  // 5) selective scan
  scan_kernel<<<BB * (DD / 256), 256, 0, stream>>>(dtbuf, u_scan, ssm, A_log, ybuf);

  // 6) gating + token-major bf16
  {
    const size_t total = (size_t)BB * DD * LL;
    gate_kernel<<<(unsigned)((total + 255) / 256), 256, 0, stream>>>(
        ybuf, u_scan, D_param, proj, y_tok);
  }

  // 7) out_proj: out[M,H] = y_tok[M,D] x w_out[H,D]^T (row-major epilogue)
  gemm_bf16_wmma<0><<<dim3(HH / 128, MM / 128), 256, 0, stream>>>(
      y_tok, wo_bf, out, MM, HH, DD, nullptr, nullptr, LL, HH);
}
